// VAE_LE_DP_26001732010261
// MI455X (gfx1250) — matrix operand order repaired and checked by reading, not compiled, not run on hardware
//
#include <hip/hip_runtime.h>
#include <hip/hip_bf16.h>
#include <math.h>
#include <cstdint>

typedef _Float16 f16;
typedef _Float16 v16h __attribute__((ext_vector_type(16)));
typedef float    v8f  __attribute__((ext_vector_type(8)));
typedef float    v2f  __attribute__((ext_vector_type(2)));
typedef unsigned uv4  __attribute__((ext_vector_type(4)));
typedef int      iv4  __attribute__((ext_vector_type(4)));
typedef int      iv8  __attribute__((ext_vector_type(8)));

union PKU { unsigned u; f16 h[2]; };

#if __has_builtin(__builtin_amdgcn_wmma_f32_16x16x4_f32)
#define HAVE_WMMA4 1
#else
#define HAVE_WMMA4 0
#endif

// ---------------- problem constants ----------------
constexpr int   C_N     = 2097152;
constexpr int   C_BLK   = 1024;
constexpr int   C_LF    = 41;
constexpr int   C_LB    = 41;
constexpr int   C_M     = 64;
constexpr int   C_PAD   = 20;           // (LF-1)/2
constexpr int   C_SPAN  = 2048;         // SPS*BLK
constexpr int   C_NBLK  = 1023;         // ceil((N-2*PAD-SPAN)/SPAN)
constexpr int   C_LOFF  = 20;           // (LB-1)/2
constexpr int   C_YBLEN = 2088;         // SPAN + 2*PAD
constexpr int   C_NVAL  = 2008;         // SPAN - LB + 1
constexpr int   C_NPAD  = 2016;         // NVAL padded to mult of 16 (zero tail)
constexpr float C_KS    = 1004.0f;      // (SPAN-LB+1)/SPS

constexpr int   C_YBI_P = 2104;         // f16 ybI pairs per pol (2088 + pad)
constexpr int   C_EQI_P = 2064;         // f16 eqI pairs per pol (2048 + pad)
constexpr int   PLN_YB  = 2 * 2096;     // f32 yb pair plane (floats)
constexpr int   PLN_EQ  = 2 * 2064;     // f32 EQfull pair plane
constexpr int   PLN_R   = 2 * 2096;     // f32 R pair plane (48 guard + 2048)

// ---------------- LDS layout (4-byte slots) ----------------
constexpr int OF_YSRE = 0;
constexpr int OF_YSIM = OF_YSRE + 2 * C_BLK;
constexpr int OF_VV   = OF_YSIM + 2 * C_BLK;
constexpr int OF_VPRE = OF_VV   + 2 * C_BLK;     // exclusive prefix, 1025/pol
constexpr int OF_TBUF = OF_VPRE + 2 * (C_BLK + 1);  // 12 f16-WMMA tiles 16x16 f32
constexpr int OF_TAPF = OF_TBUF + 12 * 256;
constexpr int OF_TAPB = OF_TAPF + 4 * C_LF * 2;
constexpr int OF_ADMF = OF_TAPB + 4 * C_LF * 2;
constexpr int OF_ADVF = OF_ADMF + 4 * C_LF * 2;
constexpr int OF_ADMB = OF_ADVF + 4 * C_LF * 2;
constexpr int OF_ADVB = OF_ADMB + 4 * C_LF * 2;
constexpr int OF_CRE  = OF_ADVB + 4 * C_LF * 2;
constexpr int OF_CIM  = OF_CRE + C_M;
constexpr int OF_AM   = OF_CIM + C_M;
constexpr int OF_LPM  = OF_AM  + C_M;
constexpr int OF_PM   = OF_LPM + C_M;
constexpr int OF_PW   = OF_PM  + C_M;            // |hb|^2 prefix, 4 x 42
constexpr int OF_SC   = OF_PW + 4 * 42;          // [0..1]=sum|R|^2, [2..3]=gC, [4]=1/sig2
constexpr int OF_CHS  = OF_SC + 8;               // scan chunk scratch 2 x 32
constexpr int OF_GCRE = OF_CHS + 64;             // Gc real [2][1024]
constexpr int OF_GCIM = OF_GCRE + 2 * C_BLK;
// f32 (re,im) pair planes for the f32-WMMA implicit GEMMs
constexpr int OF_YBIF = OF_GCIM + 2 * C_BLK;     // [pol][2096 pairs] yb
constexpr int OF_EQIF = OF_YBIF + 2 * PLN_YB;    // [pol][2064 pairs] EQfull (odd t = 0)
constexpr int OF_RIF  = OF_EQIF + 2 * PLN_EQ;    // [pol][48 guard + 2048 pairs] hEQ/R
// tap-derived B matrices (4 live cols x 96 pairs) + shared zero plane
constexpr int OF_BF1  = OF_RIF + 2 * PLN_R;      // fwd conv taps
constexpr int OF_BF3  = OF_BF1 + 4 * 192;        // hEQ conv taps
constexpr int OF_BF4  = OF_BF3 + 4 * 192;        // Gc corr taps (gC-weighted)
constexpr int OF_ZPR  = OF_BF4 + 4 * 192;        // 96 zero pairs (dead cols)
// f16-pair (uint) staging for the f16 gradient GEMMs
constexpr int U_YBI   = OF_ZPR + 192;            // [pol][2104]
constexpr int U_EQI   = U_YBI + 2 * C_YBI_P;     // [pol][2064]
constexpr int U_BTF   = U_EQI + 2 * C_EQI_P;     // [4 cols][1024]
constexpr int U_BTB   = U_BTF + 4 * C_BLK;       // [4 cols][2016]
constexpr int U_ZF16  = U_BTB + 4 * C_NPAD;      // 2016 zero uints (dead cols)
constexpr int SMEM_FLOATS = U_ZF16 + C_NPAD;     // ~266 KB (CDNA5: 320 KB/WGP)

// ---------------- TDM: 1-line tensor DMA global->LDS (ISA ch.8 D# layout) ----------
__device__ __forceinline__ void tdm_load_1d(const void* gsrc, unsigned ldsByteOff, int ndw) {
#if __has_builtin(__builtin_amdgcn_tensor_load_to_lds)
  unsigned long long ga = (unsigned long long)(uintptr_t)gsrc;
  uv4 g0 = { 1u,
             ldsByteOff,
             (unsigned)(ga & 0xFFFFFFFFu),
             (unsigned)((ga >> 32) & 0x01FFFFFFu) | (2u << 30) };
  iv8 g1 = { (int)(2u << 16),
             (int)((unsigned)ndw << 16),
             (int)((((unsigned)ndw) >> 16) | (1u << 16)),
             (int)((unsigned)ndw << 16),
             1,
             ndw,
             0,
             0 };
  iv4 gz = { 0, 0, 0, 0 };
#if defined(__clang_major__) && (__clang_major__ >= 23)
  iv8 gz8 = { 0, 0, 0, 0, 0, 0, 0, 0 };
  __builtin_amdgcn_tensor_load_to_lds(g0, g1, gz, gz, gz8, 0);
#else
  __builtin_amdgcn_tensor_load_to_lds(g0, g1, gz, gz, 0);
#endif
#endif
}

// ---------------- f32 WMMA (16x16x4) wrapper, lane-emulated fallback ----------------
__device__ __forceinline__ v8f wmma4(v2f a, v2f b, v8f c, int lane) {
#if HAVE_WMMA4
  (void)lane;
  return __builtin_amdgcn_wmma_f32_16x16x4_f32(false, a, false, b, (short)0, c,
                                               false, false);
#else
  int hl = lane >> 4, nl = lane & 15;
#pragma unroll
  for (int r = 0; r < 8; ++r) {
    int M = r + 8 * hl;
    float a0r = __shfl(a[0], M),      a0i = __shfl(a[1], M);
    float a1r = __shfl(a[0], M + 16), a1i = __shfl(a[1], M + 16);
    float b0r = __shfl(b[0], nl),      b0i = __shfl(b[1], nl);
    float b1r = __shfl(b[0], nl + 16), b1i = __shfl(b[1], nl + 16);
    c[r] += a0r * b0r + a0i * b0i + a1r * b1r + a1i * b1i;
  }
  return c;
#endif
}

// ------- unified f32 implicit-GEMM conv: D[16 rows][16 cols] over K=192 --------
// PHASE 0: y_symb[p][s]  = sum_{i2,j} yb[i2][2s+j]  * hf[2p+i2][j]   (rows = s)
// PHASE 1: hEQ[p'][n]    = sum_{i,j}  EQfull[i][n+j]* hb[2p'+i][j]   (rows = n)
// PHASE 2: Gc[i][s]      = sum_{p',j} R[p'][2s-j]   * 2 gC[p'] hb[2p'+i][j]
// K pair q = chan*48 + j (taps zero-padded 41..47); A = staged f32 (re,im) pairs.
template <int PHASE>
__device__ __forceinline__ void conv_gemm_f32(float* sm, int tileb, int lane,
                                              int ib, float* __restrict__ out_g) {
  const int col = lane & 15;
  const int hl  = lane >> 4;
  constexpr int aPlane = (PHASE == 0) ? OF_YBIF : (PHASE == 1) ? OF_EQIF : OF_RIF;
  constexpr int aStr   = (PHASE == 0) ? PLN_YB : (PHASE == 1) ? PLN_EQ : PLN_R;
  constexpr int rowMul = (PHASE == 1) ? 1 : 2;
  constexpr int xOff   = (PHASE == 2) ? 48 : 0;
  constexpr int jSign  = (PHASE == 2) ? -1 : 1;
  constexpr int bArr   = (PHASE == 0) ? OF_BF1 : (PHASE == 1) ? OF_BF3 : OF_BF4;
  const int bOff = (col < 4) ? (bArr + col * 192) : OF_ZPR;   // dead cols -> zeros
  const int arow = rowMul * (tileb + col) + xOff;
  v8f acc = {0.f, 0.f, 0.f, 0.f, 0.f, 0.f, 0.f, 0.f};
  for (int kc = 0; kc < 48; ++kc) {
    int q  = 2 * kc + hl;
    int ch = (q >= 48) ? 1 : 0;
    int j  = q - 48 * ch;
    int jc = (j > 40) ? 40 : j;            // pad taps read valid data, B is zero
    int ax = aPlane + ch * aStr + 2 * (arow + jSign * jc);
    v2f av; av[0] = sm[ax]; av[1] = sm[ax + 1];
    int bx = bOff + 2 * q;
    v2f bv; bv[0] = sm[bx]; bv[1] = sm[bx + 1];
    acc = wmma4(av, bv, acc, lane);
  }
  if (col < 4) {
    int p = col >> 1, part = col & 1;
#pragma unroll
    for (int r = 0; r < 8; ++r) {
      int rr = tileb + r + 8 * hl;         // f32 C/D lane layout (ISA 7.12.2)
      float v = acc[r];
      if constexpr (PHASE == 0) {
        sm[(part ? OF_YSIM : OF_YSRE) + p * C_BLK + rr] = v;
        size_t o = (size_t)p * ((size_t)C_NBLK * C_BLK * 2)
                 + ((size_t)ib * C_BLK + rr) * 2 + part;
        out_g[o] = v;
      } else if constexpr (PHASE == 1) {
        sm[OF_RIF + p * PLN_R + 2 * (48 + rr) + part] = v;
      } else {
        sm[(part ? OF_GCIM : OF_GCRE) + p * C_BLK + rr] = v;
      }
    }
  }
}

// ---------------- f16 WMMA tap-gradient GEMM (one 16x16 tile per wave) ------------
// GEMM 0/1 : gtf   T[ch][j]  = sum_s conj(gy[p][s]) * yb[i2][2s+j]     (K = 2*1024)
// GEMM 2/3 : gtb   G2[ch][j] = sum_n conj(R[p'][n]) * EQfull[i][n+j]   (K = 2*2016)
template <int GEMM>
__device__ __forceinline__ void tap_gemm_job(float* sm, int wave, int lane) {
  constexpr bool TF  = (GEMM < 2);
  constexpr int  POL = TF ? GEMM : (GEMM - 2);
  constexpr int  NCH = TF ? (2 * C_BLK) / 32 : (2 * C_NPAD) / 32;
  const unsigned* smU = (const unsigned*)sm;
  const int tile = wave - GEMM * 3;
  const int j0   = tile * 16;
  const int mrow = lane & 15;
  const int kh   = lane >> 4;
  const int arow = j0 + mrow;                  // rows 41..47 hit zero padding
  const unsigned* aBase = TF ? (smU + U_YBI + POL * C_YBI_P)
                             : (smU + U_EQI + POL * C_EQI_P);
  const unsigned* bBase = (mrow < 4)
      ? (TF ? (smU + U_BTF + mrow * C_BLK) : (smU + U_BTB + mrow * C_NPAD))
      : (smU + U_ZF16);                        // dead cols -> zero plane
  v8f acc = {0.f, 0.f, 0.f, 0.f, 0.f, 0.f, 0.f, 0.f};
  for (int kc = 0; kc < NCH; ++kc) {
    const int K0 = kc << 5;
    union { v16h h; unsigned u[8]; } a, b;
    const int s0 = (K0 >> 1) + kh * 4;
#pragma unroll
    for (int g = 0; g < 4; ++g) {
      if constexpr (TF) {
        a.u[g]     = aBase[2 * (s0 + g) + arow];
        a.u[4 + g] = aBase[2 * (s0 + 8 + g) + arow];
      } else {
        a.u[g]     = aBase[(s0 + g) + arow];
        a.u[4 + g] = aBase[(s0 + 8 + g) + arow];
      }
    }
    const int b0 = (K0 >> 1) + kh * 4;
#pragma unroll
    for (int g = 0; g < 8; ++g) b.u[g] = bBase[b0 + g + ((g >> 2) << 2)];
    acc = __builtin_amdgcn_wmma_f32_16x16x32_f16(
        false, a.h, false, b.h, (short)0, acc, false, false);
  }
#pragma unroll
  for (int r8 = 0; r8 < 8; ++r8) {
    int md = r8 + kh * 8;
    sm[OF_TBUF + wave * 256 + md * 16 + (lane & 15)] = acc[r8];
  }
}

__device__ __forceinline__ void adam_step(float* sm, int poff, int moff, int voff,
                                          int q0, float g, float bc1, float bc2) {
  float m = 0.9f   * sm[moff + q0] + 0.1f   * g;
  float v = 0.999f * sm[voff + q0] + 0.001f * g * g;
  sm[moff + q0] = m;
  sm[voff + q0] = v;
  sm[poff + q0] -= 0.005f * (m / bc1) / (sqrtf(v / bc2) + 1e-8f);
}

// ---------------- persistent single-WGP kernel ----------------
extern __shared__ float smem[];

__global__ void __launch_bounds__(1024, 1)
VAE_LE_DP_26001732010261_kernel(const float* __restrict__ yre_g,
                                const float* __restrict__ yim_g,
                                const float* __restrict__ tf_g,
                                const float* __restrict__ tb_g,
                                const float* __restrict__ cre_g,
                                const float* __restrict__ cim_g,
                                const float* __restrict__ ps_g,
                                const float* __restrict__ ns_g,
                                float* __restrict__ out_g) {
  float* sm = smem;
  unsigned* smU = (unsigned*)sm;
  const int tid = threadIdx.x;

  // ---- one-time setup: constants, taps, Adam state, zero guard/pad regions ----
  if (tid < C_M) {
    float cr = cre_g[tid], ci = cim_g[tid];
    sm[OF_CRE + tid] = cr;
    sm[OF_CIM + tid] = ci;
    sm[OF_AM  + tid] = cr * cr + ci * ci;
    float pv = ps_g[tid];
    sm[OF_PM  + tid] = pv;
    sm[OF_LPM + tid] = logf(pv);
  }
  if (tid < 4 * C_LF * 2) {
    sm[OF_TAPF + tid] = tf_g[tid];
    sm[OF_TAPB + tid] = tb_g[tid];
    sm[OF_ADMF + tid] = 0.0f;  sm[OF_ADVF + tid] = 0.0f;
    sm[OF_ADMB + tid] = 0.0f;  sm[OF_ADVB + tid] = 0.0f;
  }
  if (tid == 0) {
    float nsv = ns_g[0];
    sm[OF_SC + 4] = 1.0f / (nsv * nsv);
  }
  for (int t = tid; t < 2 * PLN_YB; t += 1024) sm[OF_YBIF + t] = 0.0f;
  for (int t = tid; t < 2 * PLN_EQ; t += 1024) sm[OF_EQIF + t] = 0.0f;
  for (int t = tid; t < 2 * PLN_R;  t += 1024) sm[OF_RIF  + t] = 0.0f;
  if (tid < 192) sm[OF_ZPR + tid] = 0.0f;
  for (int t = tid; t < 2 * C_YBI_P; t += 1024) smU[U_YBI + t] = 0u;
  for (int t = tid; t < 2 * C_EQI_P; t += 1024) smU[U_EQI + t] = 0u;
  for (int t = tid; t < C_NPAD; t += 1024) smU[U_ZF16 + t] = 0u;
  __syncthreads();

  const float inv_sig2 = sm[OF_SC + 4];

  for (int ib = 0; ib < C_NBLK; ++ib) {
    // ---------- P0a: TDM async tensor DMA of block input into LDS ----------
#if __has_builtin(__builtin_amdgcn_tensor_load_to_lds)
    if (tid < 32) {
      unsigned ldsBase = __builtin_amdgcn_groupstaticsize();
      for (int p = 0; p < 2; ++p) {
        size_t g = (size_t)p * C_N + (size_t)ib * C_SPAN;
        tdm_load_1d(yre_g + g, ldsBase + (OF_YBIF + p * PLN_YB) * 4u, C_YBLEN);
        tdm_load_1d(yim_g + g, ldsBase + (OF_YBIF + p * PLN_YB + C_YBLEN) * 4u, C_YBLEN);
      }
#if __has_builtin(__builtin_amdgcn_s_wait_tensorcnt)
      __builtin_amdgcn_s_wait_tensorcnt((short)0);
#endif
    }
    __syncthreads();
#endif

    // ---------- P0b: coalesced load (authoritative) + pair staging + tap B ----------
    for (int t = tid; t < 2 * C_YBLEN; t += 1024) {
      int p = t / C_YBLEN, x = t - p * C_YBLEN;
      size_t g = (size_t)p * C_N + (size_t)ib * C_SPAN + x;
      float vr = yre_g[g], vi = yim_g[g];
      sm[OF_YBIF + p * PLN_YB + 2 * x]     = vr;
      sm[OF_YBIF + p * PLN_YB + 2 * x + 1] = vi;
      PKU pk; pk.h[0] = (f16)vr; pk.h[1] = (f16)vi;
      smU[U_YBI + p * C_YBI_P + x] = pk.u;
    }
    if (tid < 2 * 4 * 96) {                      // build BF1 (hf) / BF3 (hb)
      int which = tid / 384;
      int r = tid - which * 384;
      int c = r / 96, q = r - c * 96;            // c = (p,part), q = (chan,j)
      int chn = (q >= 48) ? 1 : 0, j = q - 48 * chn;
      float b0 = 0.0f, b1 = 0.0f;
      if (j <= 40) {
        int tap = (which ? OF_TAPB : OF_TAPF) + ((2 * (c >> 1) + chn) * C_LF + j) * 2;
        float hre = sm[tap], him = sm[tap + 1];
        if ((c & 1) == 0) { b0 = hre; b1 = -him; }   // real-part channel
        else              { b0 = him; b1 = hre;  }   // imag-part channel
      }
      int base = (which ? OF_BF3 : OF_BF1) + c * 192 + 2 * q;
      sm[base] = b0; sm[base + 1] = b1;
    }
    if (tid < 4) sm[OF_SC + tid] = 0.0f;
    if (ib + 1 < C_NBLK) {                       // cacheline prefetch of next block
      constexpr int NL = (C_YBLEN + 31) / 32;
      for (int t = tid; t < 2 * NL; t += 1024) {
        int p = t / NL, x = (t - p * NL) * 32;
        if (x < C_YBLEN) {
          size_t g = (size_t)p * C_N + (size_t)(ib + 1) * C_SPAN + x;
          __builtin_prefetch(&yre_g[g], 0, 1);   // -> global_prefetch_b8
          __builtin_prefetch(&yim_g[g], 0, 1);
        }
      }
    }
    __syncthreads();

    // ---------- P1W: forward butterfly conv as f32 WMMA (64 tiles) ----------
    {
      int wave = tid >> 5, lane = tid & 31;
      for (int tb2 = wave; tb2 < 64; tb2 += 32)
        conv_gemm_f32<0>(sm, tb2 * 16, lane, ib, out_g);
    }
    __syncthreads();

    // ---------- P2: demap forward -> EQ, V (q never materialized) ----------
    for (int idx = tid; idx < 2 * C_BLK; idx += 1024) {
      int p = idx >> 10, s = idx & (C_BLK - 1);
      float are = sm[OF_YSRE + idx], aim = sm[OF_YSIM + idx];
      float zmax = -3.4e38f;
      for (int m = 0; m < C_M; ++m) {
        float dre = are - sm[OF_CRE + m], dim = aim - sm[OF_CIM + m];
        float z = -(dre * dre + dim * dim) * inv_sig2 + sm[OF_LPM + m];
        zmax = fmaxf(zmax, z);
      }
      float Sx = 0.0f;
      for (int m = 0; m < C_M; ++m) {
        float dre = are - sm[OF_CRE + m], dim = aim - sm[OF_CIM + m];
        float z = -(dre * dre + dim * dim) * inv_sig2 + sm[OF_LPM + m];
        Sx += __expf(z - zmax);
      }
      float invS = 1.0f / Sx;
      float eqre = 0.0f, eqim = 0.0f, eq2 = 0.0f;
      for (int m = 0; m < C_M; ++m) {
        float dre = are - sm[OF_CRE + m], dim = aim - sm[OF_CIM + m];
        float z = -(dre * dre + dim * dim) * inv_sig2 + sm[OF_LPM + m];
        float q = __expf(z - zmax) * invS;
        eqre += q * sm[OF_CRE + m];
        eqim += q * sm[OF_CIM + m];
        eq2  += q * sm[OF_AM  + m];
      }
      sm[OF_EQIF + p * PLN_EQ + 4 * s]     = eqre;   // pair at even t=2s
      sm[OF_EQIF + p * PLN_EQ + 4 * s + 1] = eqim;
      sm[OF_VV   + idx] = eq2 - (eqre * eqre + eqim * eqim);
      PKU pe; pe.h[0] = (f16)eqre; pe.h[1] = (f16)eqim;
      smU[U_EQI + p * C_EQI_P + 2 * s] = pe.u;
    }
    __syncthreads();

    // ---------- P2.5: exclusive prefix scan of V (per pol) + |hb|^2 prefix ----------
    if (tid < 64) {
      int i = tid >> 5, c = tid & 31;
      float ssum = 0.0f;
      for (int k2 = 0; k2 < 32; ++k2) ssum += sm[OF_VV + i * C_BLK + c * 32 + k2];
      sm[OF_CHS + tid] = ssum;
    }
    __syncthreads();
    if (tid < 2) {
      float run = 0.0f;
      for (int c = 0; c < 32; ++c) {
        float v = sm[OF_CHS + tid * 32 + c];
        sm[OF_CHS + tid * 32 + c] = run;
        run += v;
      }
      sm[OF_VPRE + tid * (C_BLK + 1) + C_BLK] = run;
    }
    if (tid >= 4 && tid < 8) {
      int c = tid - 4;
      float run = 0.0f;
      sm[OF_PW + c * 42 + 0] = 0.0f;
      for (int j = 0; j < C_LB; ++j) {
        float hre = sm[OF_TAPB + (c * C_LF + j) * 2];
        float him = sm[OF_TAPB + (c * C_LF + j) * 2 + 1];
        run += hre * hre + him * him;
        sm[OF_PW + c * 42 + j + 1] = run;
      }
    }
    __syncthreads();
    if (tid < 64) {
      int i = tid >> 5, c = tid & 31;
      float run = sm[OF_CHS + tid];
      for (int k2 = 0; k2 < 32; ++k2) {
        sm[OF_VPRE + i * (C_BLK + 1) + c * 32 + k2] = run;
        run += sm[OF_VV + i * C_BLK + c * 32 + k2];
      }
    }
    __syncthreads();

    // ---------- P3W: hEQ conv as f32 WMMA (126 tiles) -> RIF ----------
    {
      int wave = tid >> 5, lane = tid & 31;
      for (int tb2 = wave; tb2 < 126; tb2 += 32)
        conv_gemm_f32<1>(sm, tb2 * 16, lane, ib, out_g);
    }
    __syncthreads();

    // ---------- P3b: R = hEQ - yc in place ; stage Btb f16 ; sum|R|^2 ----------
    {
      float r2acc0 = 0.0f, r2acc1 = 0.0f;
      for (int idx = tid; idx < 2 * C_NPAD; idx += 1024) {
        int pp = idx / C_NPAD, n = idx - pp * C_NPAD;
        int rbase = OF_RIF + pp * PLN_R + 2 * (48 + n);
        if (n >= C_NVAL) {
          sm[rbase] = 0.0f; sm[rbase + 1] = 0.0f;
          smU[U_BTB + (2 * pp) * C_NPAD + n]     = 0u;
          smU[U_BTB + (2 * pp + 1) * C_NPAD + n] = 0u;
          continue;
        }
        int ybase = OF_YBIF + pp * PLN_YB + 2 * (C_PAD + C_LOFF + n);
        float rre = sm[rbase]     - sm[ybase];
        float rim = sm[rbase + 1] - sm[ybase + 1];
        sm[rbase] = rre; sm[rbase + 1] = rim;
        PKU c0; c0.h[0] = (f16)rre;    c0.h[1] = (f16)rim;
        PKU c1; c1.h[0] = (f16)(-rim); c1.h[1] = (f16)rre;
        smU[U_BTB + (2 * pp) * C_NPAD + n]     = c0.u;
        smU[U_BTB + (2 * pp + 1) * C_NPAD + n] = c1.u;
        float r2 = rre * rre + rim * rim;
        if (pp == 0) r2acc0 += r2; else r2acc1 += r2;
      }
      if (r2acc0 != 0.0f) atomicAdd(&sm[OF_SC + 0], r2acc0);  // ds_add_f32
      if (r2acc1 != 0.0f) atomicAdd(&sm[OF_SC + 1], r2acc1);
    }
    __syncthreads();

    // ---------- P3.5: C[p] = sum|R|^2 + sum Vterm ; gC = Ks/(C+eps) ----------
    if (tid < 2) {
      int pp = tid;
      float vt = 0.0f;
      for (int i = 0; i < 2; ++i)
        for (int j = 0; j < C_LB; ++j) {
          float hre = sm[OF_TAPB + ((2 * pp + i) * C_LF + j) * 2];
          float him = sm[OF_TAPB + ((2 * pp + i) * C_LF + j) * 2 + 1];
          float W = sm[OF_VPRE + i * (C_BLK + 1) + ((j + C_NVAL + 1) >> 1)]
                  - sm[OF_VPRE + i * (C_BLK + 1) + ((j + 1) >> 1)];
          vt += (hre * hre + him * him) * W;
        }
      float Cp = sm[OF_SC + pp] + vt;
      sm[OF_SC + 2 + pp] = C_KS / (Cp + 1e-8f);
    }
    __syncthreads();

    // ---------- P3.6: build BF4 = 2*gC[p']*hb combos for the Gc GEMM ----------
    if (tid < 4 * 96) {
      int c = tid / 96, q = tid - c * 96;        // c = (i,part), q = (p',j)
      int pp = (q >= 48) ? 1 : 0, j = q - 48 * pp;
      float b0 = 0.0f, b1 = 0.0f;
      if (j <= 40) {
        float gc2 = 2.0f * sm[OF_SC + 2 + pp];
        int tap = OF_TAPB + ((2 * pp + (c >> 1)) * C_LF + j) * 2;
        float hre = sm[tap], him = sm[tap + 1];
        if ((c & 1) == 0) { b0 = gc2 * hre; b1 = gc2 * him; }   // Re(conj(R)h)
        else              { b0 = gc2 * him; b1 = -gc2 * hre; }  // Im(conj(R)h)
      }
      sm[OF_BF4 + c * 192 + 2 * q]     = b0;
      sm[OF_BF4 + c * 192 + 2 * q + 1] = b1;
    }
    __syncthreads();

    // ---------- P4W: Gc correlation as f32 WMMA (64 tiles) ----------
    {
      int wave = tid >> 5, lane = tid & 31;
      for (int tb2 = wave; tb2 < 64; tb2 += 32)
        conv_gemm_f32<2>(sm, tb2 * 16, lane, ib, out_g);
    }
    __syncthreads();

    // ---------- P5: backward demap -> gy -> Btf f16 staging ----------
    for (int idx = tid; idx < 2 * C_BLK; idx += 1024) {
      int p = idx >> 10, s = idx & (C_BLK - 1);
      int t = 2 * s;
      float gr = sm[OF_GCRE + idx];
      float gi = sm[OF_GCIM + idx];
      float hcv = 0.0f;
      for (int pp = 0; pp < 2; ++pp) {
        float gc = sm[OF_SC + 2 + pp];
        int jlo = (t - (C_NVAL - 1)) > 0 ? (t - (C_NVAL - 1)) : 0;
        int jhi = t < (C_LB - 1) ? t : (C_LB - 1);
        int c0 = 2 * pp + p;
        hcv += gc * (sm[OF_PW + c0 * 42 + jhi + 1] - sm[OF_PW + c0 * 42 + jlo]);
      }
      float are = sm[OF_YSRE + idx], aim = sm[OF_YSIM + idx];
      float eqre = sm[OF_EQIF + p * PLN_EQ + 4 * s];
      float eqim = sm[OF_EQIF + p * PLN_EQ + 4 * s + 1];
      bool interior = (s >= C_LOFF) && (s < C_BLK - C_LOFF);
      float zmax = -3.4e38f;
      for (int m = 0; m < C_M; ++m) {
        float dre = are - sm[OF_CRE + m], dim = aim - sm[OF_CIM + m];
        float z = -(dre * dre + dim * dim) * inv_sig2 + sm[OF_LPM + m];
        zmax = fmaxf(zmax, z);
      }
      float Sx = 0.0f;
      for (int m = 0; m < C_M; ++m) {
        float dre = are - sm[OF_CRE + m], dim = aim - sm[OF_CIM + m];
        float z = -(dre * dre + dim * dim) * inv_sig2 + sm[OF_LPM + m];
        Sx += __expf(z - zmax);
      }
      float invS = 1.0f / Sx;
      float S1 = 0.0f;
      for (int m = 0; m < C_M; ++m) {
        float cr = sm[OF_CRE + m], ci = sm[OF_CIM + m];
        float dre = are - cr, dim = aim - ci;
        float z = -(dre * dre + dim * dim) * inv_sig2 + sm[OF_LPM + m];
        float q = __expf(z - zmax) * invS;
        float dq = gr * cr - gi * ci
                 + hcv * (sm[OF_AM + m] - 2.0f * (eqre * cr + eqim * ci));
        if (interior) {
          float qp = q / sm[OF_PM + m];
          float lv = qp + 1e-14f;
          dq += logf(lv) + qp / lv;
        }
        S1 += q * dq;
      }
      float gyr = 0.0f, gyi = 0.0f;
      for (int m = 0; m < C_M; ++m) {
        float cr = sm[OF_CRE + m], ci = sm[OF_CIM + m];
        float dre = are - cr, dim = aim - ci;
        float z = -(dre * dre + dim * dim) * inv_sig2 + sm[OF_LPM + m];
        float q = __expf(z - zmax) * invS;
        float dq = gr * cr - gi * ci
                 + hcv * (sm[OF_AM + m] - 2.0f * (eqre * cr + eqim * ci));
        if (interior) {
          float qp = q / sm[OF_PM + m];
          float lv = qp + 1e-14f;
          dq += logf(lv) + qp / lv;
        }
        float dz = q * (dq - S1);
        gyr += dz * cr;
        gyi += dz * ci;
      }
      float gre = 2.0f * inv_sig2 * gyr;
      float gim = 2.0f * inv_sig2 * gyi;
      PKU c0; c0.h[0] = (f16)gre;    c0.h[1] = (f16)gim;
      PKU c1; c1.h[0] = (f16)(-gim); c1.h[1] = (f16)gre;
      smU[U_BTF + (2 * p) * C_BLK + s]     = c0.u;
      smU[U_BTF + (2 * p + 1) * C_BLK + s] = c1.u;
    }
    __syncthreads();

    // ---------- P6: tap-gradient correlations as f16 WMMA GEMMs ----------
    {
      int wave = tid >> 5;
      if (wave < 12) {
        int lane = tid & 31;
        int gemm = wave / 3;
        if (gemm == 0)      tap_gemm_job<0>(sm, wave, lane);
        else if (gemm == 1) tap_gemm_job<1>(sm, wave, lane);
        else if (gemm == 2) tap_gemm_job<2>(sm, wave, lane);
        else                tap_gemm_job<3>(sm, wave, lane);
      }
    }
    __syncthreads();

    // ---------- P7: assemble gradients, Adam update of taps (kept in LDS) ----------
    {
      float tstep = (float)(ib + 1);
      float bc1 = 1.0f - __powf(0.9f, tstep);
      float bc2 = 1.0f - __powf(0.999f, tstep);
      if (tid < 2 * 4 * C_LF) {
        int grp = tid / (4 * C_LF);              // 0: tf grads, 1: tb grads
        int r = tid - grp * (4 * C_LF);
        int c = r / C_LF, j = r - c * C_LF;
        int tile = j >> 4, jm = j & 15;
        int q0 = (c * C_LF + j) * 2;
        if (grp == 0) {
          int p = c >> 1, i2 = c & 1;
          int job = i2 * 3 + tile;
          float Tre = sm[OF_TBUF + job * 256 + jm * 16 + 2 * p];
          float Tim = sm[OF_TBUF + job * 256 + jm * 16 + 2 * p + 1];
          adam_step(sm, OF_TAPF, OF_ADMF, OF_ADVF, q0,     Tre,  bc1, bc2);
          adam_step(sm, OF_TAPF, OF_ADMF, OF_ADVF, q0 + 1, -Tim, bc1, bc2);
        } else {
          int pp = c >> 1, i = c & 1;
          int job = (2 + i) * 3 + tile;
          float G2re = sm[OF_TBUF + job * 256 + jm * 16 + 2 * pp];
          float G2im = sm[OF_TBUF + job * 256 + jm * 16 + 2 * pp + 1];
          float Wv = sm[OF_VPRE + i * (C_BLK + 1) + ((j + C_NVAL + 1) >> 1)]
                   - sm[OF_VPRE + i * (C_BLK + 1) + ((j + 1) >> 1)];
          float hre = sm[OF_TAPB + q0], him = sm[OF_TAPB + q0 + 1];
          float gc = sm[OF_SC + 2 + pp];
          float gre = gc * ( 2.0f * G2re + 2.0f * hre * Wv);
          float gim = gc * (-2.0f * G2im + 2.0f * him * Wv);
          adam_step(sm, OF_TAPB, OF_ADMB, OF_ADVB, q0,     gre, bc1, bc2);
          adam_step(sm, OF_TAPB, OF_ADMB, OF_ADVB, q0 + 1, gim, bc1, bc2);
        }
      }
    }
    __syncthreads();   // taps updated before next block's forward pass
  }
}

// ---------------- host launcher ----------------
extern "C" void kernel_launch(void* const* d_in, const int* in_sizes, int n_in,
                              void* d_out, int out_size, void* d_ws, size_t ws_size,
                              hipStream_t stream) {
  (void)in_sizes; (void)n_in; (void)out_size; (void)d_ws; (void)ws_size;
  const float* yre = (const float*)d_in[0];
  const float* yim = (const float*)d_in[1];
  const float* tf0 = (const float*)d_in[2];
  const float* tb0 = (const float*)d_in[3];
  const float* cre = (const float*)d_in[4];
  const float* cim = (const float*)d_in[5];
  const float* ps  = (const float*)d_in[6];
  const float* ns  = (const float*)d_in[7];
  float* out = (float*)d_out;

  size_t shbytes = (size_t)SMEM_FLOATS * sizeof(float);   // ~266 KB (CDNA5: 320 KB/WGP)
  (void)hipFuncSetAttribute((const void*)VAE_LE_DP_26001732010261_kernel,
                            hipFuncAttributeMaxDynamicSharedMemorySize, (int)shbytes);
  VAE_LE_DP_26001732010261_kernel<<<dim3(1), dim3(1024), shbytes, stream>>>(
      yre, yim, tf0, tb0, cre, cim, ps, ns, out);
}
